// CrossAttention_44246753083653
// MI455X (gfx1250) — matrix operand order repaired and checked by reading, not compiled, not run on hardware
//
#include <hip/hip_runtime.h>

typedef __attribute__((ext_vector_type(16))) _Float16 v16h;
typedef __attribute__((ext_vector_type(8)))  float    v8f;

#define CC 1024
#define BB 128

// ---------------------------------------------------------------------------
// Kernel 1: pack center tap of 3x3 conv weights -> f16 [O][I]
// conv_w layout [O][I][3][3]; center element at flat*9 + 4.
// ---------------------------------------------------------------------------
__global__ __launch_bounds__(256) void pack_w_kernel(const float* __restrict__ w,
                                                     _Float16* __restrict__ wp) {
    int idx = blockIdx.x * 256 + threadIdx.x;   // 0 .. 1024*1024-1
    wp[idx] = (_Float16)w[idx * 9 + 4];
}

// ---------------------------------------------------------------------------
// Kernel 2: fused rank-1 softmax cross-attention + residual, f16 output.
// One block per batch row b. visual[b,:] cached in LDS; stable softmax via
// m = t>0 ? t*vmax : t*vmin. Each thread owns 4 rows i.
// ---------------------------------------------------------------------------
__global__ __launch_bounds__(256) void attn_kernel(const float* __restrict__ visual,
                                                   const float* __restrict__ tactile,
                                                   _Float16* __restrict__ h16) {
    __shared__ float sV[CC];
    __shared__ float sMax[256];
    __shared__ float sMin[256];
    const int b = blockIdx.x;
    const int tid = threadIdx.x;

    float4 v4 = ((const float4*)(visual + b * CC))[tid];
    ((float4*)sV)[tid] = v4;
    sMax[tid] = fmaxf(fmaxf(v4.x, v4.y), fmaxf(v4.z, v4.w));
    sMin[tid] = fminf(fminf(v4.x, v4.y), fminf(v4.z, v4.w));
    __syncthreads();
    for (int s = 128; s > 0; s >>= 1) {
        if (tid < s) {
            sMax[tid] = fmaxf(sMax[tid], sMax[tid + s]);
            sMin[tid] = fminf(sMin[tid], sMin[tid + s]);
        }
        __syncthreads();
    }
    const float vmax = sMax[0], vmin = sMin[0];
    const float L2E = 1.44269504088896f;

    float t2[4], m2[4], S0[4], S1[4];
    #pragma unroll
    for (int q = 0; q < 4; ++q) {
        int i = tid + q * 256;
        float tt = tactile[b * CC + i] * L2E;
        t2[q] = tt;
        m2[q] = (tt > 0.f) ? tt * vmax : tt * vmin;   // max_j of tt*v_j
        S0[q] = 0.f; S1[q] = 0.f;
    }
    for (int j = 0; j < CC; ++j) {
        float v = sV[j];
        #pragma unroll
        for (int q = 0; q < 4; ++q) {
            float e = __builtin_amdgcn_exp2f(__builtin_fmaf(t2[q], v, -m2[q]));
            S0[q] += e;
            S1[q] = __builtin_fmaf(v, e, S1[q]);
        }
    }
    #pragma unroll
    for (int q = 0; q < 4; ++q) {
        int i = tid + q * 256;
        float crossed = S1[q] * __builtin_amdgcn_rcpf(S0[q]);
        h16[b * CC + i] = (_Float16)(sV[i] + crossed);
    }
}

// ---------------------------------------------------------------------------
// Kernel 3: Y = H(128x1024) x Wc^T(1024x1024) via v_wmma_f32_16x16x32_f16,
// fused bias + BatchNorm(eval) + LeakyReLU(0.1).
// 8 waves/block, one 16x16 tile per wave; 512 tiles -> 64 blocks.
// A as loaded here (contiguous k, not AMD's table order): lane<16 row M=lane holds
//   K={0..15}; lane>=16 holds K={16..31} (pairs per VGPR).
// B layout (32x16): lane holds column N=lane%16, K = 16 consecutive values,
//   upper-half lanes offset by +16 -> two contiguous 16B loads.
// ---------------------------------------------------------------------------
__global__ __launch_bounds__(256) void gemm_bn_lrelu_kernel(
        const _Float16* __restrict__ A16,   // H, f16 [128][1024]
        const _Float16* __restrict__ W16,   // Wc, f16 [O=1024][I=1024]
        const float* __restrict__ bias,
        const float* __restrict__ gamma,
        const float* __restrict__ beta,
        const float* __restrict__ mean,
        const float* __restrict__ var,
        float* __restrict__ out) {
    const int wave = threadIdx.x >> 5;
    const int lane = threadIdx.x & 31;
    const int tileLin = blockIdx.x * 8 + wave;   // 0..511
    const int tileN = tileLin & 63;              // 64 tiles over O
    const int tileM = tileLin >> 6;              // 8 tiles over batch
    const int m0 = tileM * 16;
    const int n0 = tileN * 16;
    const int hi = lane >> 4;                    // lane half
    const int l  = lane & 15;

    const uint32_t* Arow = (const uint32_t*)(A16 + (m0 + l) * CC);
    const _Float16* Brow = W16 + (n0 + l) * CC + hi * 16;

    v8f c = {};
    for (int k0 = 0; k0 < CC; k0 += 32) {
        union { uint32_t u[8]; v16h h; } a;
        #pragma unroll
        for (int p = 0; p < 8; ++p) {
            int koff = 2 * p + hi * 16;   // K of half-pair p
            a.u[p] = Arow[(k0 + koff) >> 1];
        }
        union { uint4 q[2]; v16h h; } bm;
        const uint4* bp = (const uint4*)(Brow + k0);
        bm.q[0] = bp[0];
        bm.q[1] = bp[1];
        c = __builtin_amdgcn_wmma_f32_16x16x32_f16(
                /*neg_a=*/false, a.h, /*neg_b=*/false, bm.h,
                /*c_mod=*/(short)0, c, /*reuse_a=*/false, /*reuse_b=*/false);
    }

    // Epilogue: C/D layout -> per lane N = l is constant, M = r + hi*8.
    const int o   = n0 + l;
    const float cb  = bias[o];
    const float inv = gamma[o] * rsqrtf(var[o] + 1e-5f);
    const float mu  = mean[o];
    const float bt  = beta[o];
    #pragma unroll
    for (int r = 0; r < 8; ++r) {
        int m = m0 + r + hi * 8;
        float y = c[r] + cb;
        y = (y - mu) * inv + bt;
        y = (y > 0.f) ? y : 0.1f * y;
        out[m * CC + o] = y;
    }
}

// ---------------------------------------------------------------------------
extern "C" void kernel_launch(void* const* d_in, const int* in_sizes, int n_in,
                              void* d_out, int out_size, void* d_ws, size_t ws_size,
                              hipStream_t stream) {
    const float* visual  = (const float*)d_in[0];
    const float* tactile = (const float*)d_in[1];
    const float* conv_w  = (const float*)d_in[2];
    const float* conv_b  = (const float*)d_in[3];
    const float* gamma   = (const float*)d_in[4];
    const float* beta    = (const float*)d_in[5];
    const float* mean    = (const float*)d_in[6];
    const float* var     = (const float*)d_in[7];
    float* out = (float*)d_out;

    _Float16* Hf16 = (_Float16*)d_ws;                            // 256 KB
    _Float16* Wf16 = (_Float16*)((char*)d_ws + (512 * 1024));    // 2 MB

    pack_w_kernel<<<(CC * CC) / 256, 256, 0, stream>>>(conv_w, Wf16);
    attn_kernel<<<BB, 256, 0, stream>>>(visual, tactile, Hf16);
    gemm_bn_lrelu_kernel<<<(BB / 16) * (CC / 16) / 8, 256, 0, stream>>>(
        Hf16, Wf16, conv_b, gamma, beta, mean, var, out);
}
